// SimpleAttentionModel_4071628997014
// MI455X (gfx1250) — matrix operand order repaired and checked by reading, not compiled, not run on hardware
//
#include <hip/hip_runtime.h>
#include <hip/hip_bf16.h>
#include <math.h>

typedef __attribute__((ext_vector_type(4)))  _Float16 v4h;
typedef __attribute__((ext_vector_type(8)))  _Float16 v8h;
typedef __attribute__((ext_vector_type(16))) _Float16 v16h;
typedef __attribute__((ext_vector_type(8)))  float    v8f;
typedef __attribute__((__vector_size__(4 * sizeof(int)))) int vi4;  // matches builtin param

#define B_ 8
#define S_ 2048
#define E_ 512
#define H_ 2048
#define V_ 32000

#if __has_builtin(__builtin_amdgcn_global_load_async_to_lds_b128)
#define ASYNC_LDS 1
#else
#define ASYNC_LDS 0
#endif

__device__ inline void async_wait_3() {
#if __has_builtin(__builtin_amdgcn_s_wait_asynccnt)
  __builtin_amdgcn_s_wait_asynccnt(3);
#else
  asm volatile("s_wait_asynccnt 3" ::: "memory");
#endif
}
__device__ inline void async_wait_0() {
#if __has_builtin(__builtin_amdgcn_s_wait_asynccnt)
  __builtin_amdgcn_s_wait_asynccnt(0);
#else
  asm volatile("s_wait_asynccnt 0" ::: "memory");
#endif
}

// ---------------- embedding gather + f32->f16 ----------------
__global__ void k_gather(const int* __restrict__ x, const float* __restrict__ emb,
                         _Float16* __restrict__ e16) {
  int row = blockIdx.x;                       // 0..B*S-1
  long idx = (long)x[row];
  const float4* src = (const float4*)(emb + idx * E_);
  float4 f = src[threadIdx.x];                // 128 threads * 4 = 512
  v4h h;
  h[0] = (_Float16)f.x; h[1] = (_Float16)f.y;
  h[2] = (_Float16)f.z; h[3] = (_Float16)f.w;
  ((v4h*)(e16 + (long)row * E_))[threadIdx.x] = h;
}

__global__ void k_f32_to_f16(const float* __restrict__ in, _Float16* __restrict__ out, int n) {
  int i = blockIdx.x * 256 + threadIdx.x;
  if (i < n) out[i] = (_Float16)in[i];
}

// ---------------- WMMA fragment loaders from LDS tiles ----------------
// A tile 64x32 (lda=32): 16-bit A layout -> lanes 0-15 rows, K {kb..kb+7, kb+16..kb+23}
__device__ inline v16h frag_a_lds(const _Float16* T, int r0, int lane) {
  int row = r0 + (lane & 15);
  int kb  = (lane & 16) ? 8 : 0;
  const _Float16* p = T + row * 32 + kb;
  v8h lo = *(const v8h*)p;
  v8h hi = *(const v8h*)(p + 16);
  v16h r;
#pragma unroll
  for (int i = 0; i < 8; ++i) { r[i] = lo[i]; r[i + 8] = hi[i]; }
  return r;
}
// B tile 128x32 (ldb=32): lanes 0-15 cols hold K {0..7, 16..23}, lanes 16-31 hold K {8..15, 24..31}
__device__ inline v16h frag_b_lds(const _Float16* T, int c0, int lane) {
  int col = c0 + (lane & 15);
  int kb  = (lane & 16) ? 8 : 0;
  const _Float16* p = T + col * 32 + kb;
  v8h lo = *(const v8h*)p;
  v8h hi = *(const v8h*)(p + 16);
  v16h r;
#pragma unroll
  for (int i = 0; i < 8; ++i) { r[i] = lo[i]; r[i + 8] = hi[i]; }
  return r;
}

// Cooperative stage of A 64x32 (4KB) + B 128x32 (8KB) K-slices into LDS.
// 256 threads: 1 x 16B for A, 2 x 16B for B per thread (3 async ops/wave).
__device__ inline void stage_tiles(const _Float16* __restrict__ Ag,
                                   const _Float16* __restrict__ Bg,
                                   int K, int k0,
                                   _Float16* As, _Float16* Bs, int t) {
  {
    int row = t >> 2, seg = t & 3;
    const _Float16* g = Ag + (long)row * K + k0 + seg * 8;
#if ASYNC_LDS
    __builtin_amdgcn_global_load_async_to_lds_b128((vi4*)g, (vi4*)(As + t * 8), 0, 0);
#else
    *(v8h*)(As + t * 8) = *(const v8h*)g;
#endif
  }
#pragma unroll
  for (int j = 0; j < 2; ++j) {
    int idx = t + j * 256;
    int row = idx >> 2, seg = idx & 3;
    const _Float16* g = Bg + (long)row * K + k0 + seg * 8;
#if ASYNC_LDS
    __builtin_amdgcn_global_load_async_to_lds_b128((vi4*)g, (vi4*)(Bs + idx * 8), 0, 0);
#else
    *(v8h*)(Bs + idx * 8) = *(const v8h*)g;
#endif
  }
}

// ---------------- WMMA GEMM: C[M,N] = A[M,K] * B[N,K]^T ----------------
// 256 threads = 8 waves as 4(M) x 2(N); block tile 64(M) x 128(N).
// Each wave: one A-frag amortized over 4 B-frags -> 4 independent WMMAs per K-step.
// LDS double-buffered; async-to-LDS overlapped with compute when available.
template <bool OUT_F16>
__global__ void k_gemm_abT(const _Float16* __restrict__ A, const _Float16* __restrict__ Bm,
                           void* __restrict__ Cv, int N, int K,
                           long sA, long sB, long sC) {
  __shared__ _Float16 As[2][64 * 32];    // 2 x 4KB
  __shared__ _Float16 Bs[2][128 * 32];   // 2 x 8KB
  int t = threadIdx.x, lane = t & 31, wid = t >> 5;
  const _Float16* Ag = A  + blockIdx.z * sA + (long)(blockIdx.x * 64) * K;
  const _Float16* Bg = Bm + blockIdx.z * sB + (long)(blockIdx.y * 128) * K;
  int mrow = (wid >> 1) * 16;   // 0,16,32,48
  int ncol = (wid & 1) * 64;    // 0,64
  v8f acc[4] = {{}, {}, {}, {}};
  int nk = K >> 5;

  stage_tiles(Ag, Bg, K, 0, As[0], Bs[0], t);
  for (int i = 0; i < nk; ++i) {
    int cur = i & 1;
    if (i + 1 < nk) {
      stage_tiles(Ag, Bg, K, (i + 1) << 5, As[cur ^ 1], Bs[cur ^ 1], t);
#if ASYNC_LDS
      async_wait_3();   // previous slice's 3 asyncs retired; next 3 still in flight
#endif
    } else {
#if ASYNC_LDS
      async_wait_0();
#endif
    }
    __syncthreads();                           // slice i visible to all waves
    v16h a  = frag_a_lds(As[cur], mrow, lane);
    v16h b0 = frag_b_lds(Bs[cur], ncol +  0, lane);
    v16h b1 = frag_b_lds(Bs[cur], ncol + 16, lane);
    v16h b2 = frag_b_lds(Bs[cur], ncol + 32, lane);
    v16h b3 = frag_b_lds(Bs[cur], ncol + 48, lane);
    acc[0] = __builtin_amdgcn_wmma_f32_16x16x32_f16(false, a, false, b0, (short)0, acc[0], false, false);
    acc[1] = __builtin_amdgcn_wmma_f32_16x16x32_f16(false, a, false, b1, (short)0, acc[1], false, false);
    acc[2] = __builtin_amdgcn_wmma_f32_16x16x32_f16(false, a, false, b2, (short)0, acc[2], false, false);
    acc[3] = __builtin_amdgcn_wmma_f32_16x16x32_f16(false, a, false, b3, (short)0, acc[3], false, false);
    __syncthreads();                           // all reads of slice i done before overwrite
  }

  // C/D layout: lanes 0-15 -> M=0..7 (vgpr r), N=lane; lanes 16-31 -> M=8..15, N=lane-16
  int col = lane & 15;
  int rb  = (lane & 16) ? 8 : 0;
  int mg  = blockIdx.x * 64 + mrow + rb;
  int ng  = blockIdx.y * 128 + ncol;
  if constexpr (OUT_F16) {
    _Float16* C = (_Float16*)Cv + blockIdx.z * sC;
#pragma unroll
    for (int s = 0; s < 4; ++s)
#pragma unroll
      for (int r = 0; r < 8; ++r)
        C[(long)(mg + r) * N + ng + s * 16 + col] = (_Float16)acc[s][r];
  } else {
    float* C = (float*)Cv + blockIdx.z * sC;
#pragma unroll
    for (int s = 0; s < 4; ++s)
#pragma unroll
      for (int r = 0; r < 8; ++r)
        C[(long)(mg + r) * N + ng + s * 16 + col] = acc[s][r];
  }
}

// ---------------- block reductions (wave32) ----------------
__device__ inline float wave_max(float v) {
#pragma unroll
  for (int o = 16; o > 0; o >>= 1) v = fmaxf(v, __shfl_xor(v, o, 32));
  return v;
}
__device__ inline float wave_sum(float v) {
#pragma unroll
  for (int o = 16; o > 0; o >>= 1) v += __shfl_xor(v, o, 32);
  return v;
}
__device__ inline float block_max(float v, float* red) {
  int t = threadIdx.x, w = t >> 5, l = t & 31;
  v = wave_max(v);
  if (l == 0) red[w] = v;
  __syncthreads();
  if (w == 0) {
    float x = (l < 8) ? red[l] : -INFINITY;
    x = wave_max(x);
    if (l == 0) red[0] = x;
  }
  __syncthreads();
  float out = red[0];
  __syncthreads();
  return out;
}
__device__ inline float block_sum(float v, float* red) {
  int t = threadIdx.x, w = t >> 5, l = t & 31;
  v = wave_sum(v);
  if (l == 0) red[w] = v;
  __syncthreads();
  if (w == 0) {
    float x = (l < 8) ? red[l] : 0.f;
    x = wave_sum(x);
    if (l == 0) red[0] = x;
  }
  __syncthreads();
  float out = red[0];
  __syncthreads();
  return out;
}

// ---------------- softmax column sums (deterministic, per-group partials) -------
__global__ void k_colsum(const float* __restrict__ scores, float* __restrict__ wpart) {
  __shared__ float wloc[S_];   // 8 KB
  __shared__ float red[8];
  int g = blockIdx.x, b = blockIdx.y, t = threadIdx.x;
  for (int i = t; i < S_; i += 256) wloc[i] = 0.f;
  __syncthreads();
  const float* Sb = scores + ((long)b * S_ + (long)g * 64) * S_;
  for (int r = 0; r < 64; ++r) {
    const float* row = Sb + (long)r * S_;
    float v[8], p[8];
    float m = -INFINITY;
#pragma unroll
    for (int i = 0; i < 8; ++i) { v[i] = row[t + i * 256]; m = fmaxf(m, v[i]); }
    m = block_max(m, red);
    float s = 0.f;
#pragma unroll
    for (int i = 0; i < 8; ++i) { p[i] = __expf(v[i] - m); s += p[i]; }
    s = block_sum(s, red);
    float inv = 1.f / s;
#pragma unroll
    for (int i = 0; i < 8; ++i) wloc[t + i * 256] += p[i] * inv;  // thread-private slots
  }
  __syncthreads();
  float* wp = wpart + ((long)b * 32 + g) * S_;
  for (int i = t; i < S_; i += 256) wp[i] = wloc[i];
}

__global__ void k_wreduce(const float* __restrict__ wpart, float* __restrict__ w) {
  int b = blockIdx.x;
  for (int k = threadIdx.x; k < S_; k += 256) {
    float s = 0.f;
#pragma unroll 4
    for (int g = 0; g < 32; ++g) s += wpart[((long)b * 32 + g) * S_ + k];
    w[b * S_ + k] = s;
  }
}

// ---------------- v_out[b,d] = sum_k w[b,k] * v[b,k,d] ----------------
__global__ void k_vout(const float* __restrict__ w, const _Float16* __restrict__ vmat,
                       float* __restrict__ vout) {
  int b = blockIdx.x, t = threadIdx.x;
  const _Float16* Vb = vmat + (long)b * S_ * E_;
  const float* wb = w + b * S_;
  float a0 = 0.f, a1 = 0.f;
  for (int k = 0; k < S_; ++k) {
    float wk = wb[k];
    a0 += wk * (float)Vb[(long)k * E_ + t];
    a1 += wk * (float)Vb[(long)k * E_ + t + 256];
  }
  vout[b * E_ + t] = a0;
  vout[b * E_ + t + 256] = a1;
}

// ---------------- h = relu(v_out @ W1^T + b1)  (M=8: bandwidth-bound) -----------
__global__ void k_mlp1(const float* __restrict__ vout, const float* __restrict__ W1,
                       const float* __restrict__ b1, float* __restrict__ h) {
  __shared__ float vsh[E_];
  int b = blockIdx.y;
  int hh = blockIdx.x * 256 + threadIdx.x;
  for (int i = threadIdx.x; i < E_; i += 256) vsh[i] = vout[b * E_ + i];
  __syncthreads();
  const float4* Wr = (const float4*)(W1 + (long)hh * E_);
  float acc = 0.f;
#pragma unroll 4
  for (int j = 0; j < E_ / 4; ++j) {
    float4 f = Wr[j];
    acc += f.x * vsh[4 * j] + f.y * vsh[4 * j + 1] + f.z * vsh[4 * j + 2] + f.w * vsh[4 * j + 3];
  }
  acc += b1[hh];
  h[b * H_ + hh] = fmaxf(acc, 0.f);
}

// ---------------- logits = h @ W2^T + b2 : stream 262MB of W2 exactly once ------
__global__ void k_logits(const float* __restrict__ h, const float* __restrict__ W2,
                         const float* __restrict__ b2, float* __restrict__ logits) {
  __shared__ _Float16 hsh[B_ * H_];   // 32 KB
  for (int i = threadIdx.x; i < B_ * H_; i += 256) hsh[i] = (_Float16)h[i];
  __syncthreads();
  int v = blockIdx.x * 256 + threadIdx.x;   // 125*256 == 32000, always in range
  const float4* Wr = (const float4*)(W2 + (long)v * H_);
  float acc[B_];
#pragma unroll
  for (int b = 0; b < B_; ++b) acc[b] = 0.f;
  for (int j = 0; j < H_ / 4; ++j) {
    float4 f = Wr[j];
    __builtin_prefetch(Wr + j + 16, 0, 0);   // stream W2 ahead
#pragma unroll
    for (int b = 0; b < B_; ++b) {
      const _Float16* hp = &hsh[b * H_ + 4 * j];
      acc[b] += f.x * (float)hp[0] + f.y * (float)hp[1] +
                f.z * (float)hp[2] + f.w * (float)hp[3];
    }
  }
  float bb = b2[v];
#pragma unroll
  for (int b = 0; b < B_; ++b) logits[(long)b * V_ + v] = acc[b] + bb;
}

// ---------------- final row softmax over [8, 32000] ----------------
__global__ void k_softmax(const float* __restrict__ logits, float* __restrict__ out) {
  __shared__ float red[8];
  int b = blockIdx.x, t = threadIdx.x;
  const float* L = logits + (long)b * V_;
  float m = -INFINITY;
  for (int i = t; i < V_; i += 256) m = fmaxf(m, L[i]);
  m = block_max(m, red);
  float s = 0.f;
  for (int i = t; i < V_; i += 256) s += __expf(L[i] - m);
  s = block_sum(s, red);
  float inv = 1.f / s;
  float* O = out + (long)b * V_;
  for (int i = t; i < V_; i += 256) O[i] = __expf(L[i] - m) * inv;
}

// ---------------- launcher ----------------
extern "C" void kernel_launch(void* const* d_in, const int* in_sizes, int n_in,
                              void* d_out, int out_size, void* d_ws, size_t ws_size,
                              hipStream_t stream) {
  const int*   x   = (const int*)d_in[0];
  const float* emb = (const float*)d_in[1];
  const float* Wq  = (const float*)d_in[2];
  const float* Wk  = (const float*)d_in[3];
  const float* Wv  = (const float*)d_in[4];
  const float* W1  = (const float*)d_in[5];
  const float* b1  = (const float*)d_in[6];
  const float* W2  = (const float*)d_in[7];
  const float* b2  = (const float*)d_in[8];

  char* ws = (char*)d_ws;
  size_t off = 0;
  auto carve = [&](size_t bytes) { void* p = ws + off; off = (off + bytes + 255) & ~(size_t)255; return p; };
  _Float16* e16    = (_Float16*)carve((size_t)B_ * S_ * E_ * 2);
  _Float16* q16    = (_Float16*)carve((size_t)B_ * S_ * E_ * 2);
  _Float16* kk16   = (_Float16*)carve((size_t)B_ * S_ * E_ * 2);
  _Float16* v16m   = (_Float16*)carve((size_t)B_ * S_ * E_ * 2);
  _Float16* wq16   = (_Float16*)carve((size_t)E_ * E_ * 2);
  _Float16* wk16   = (_Float16*)carve((size_t)E_ * E_ * 2);
  _Float16* wv16   = (_Float16*)carve((size_t)E_ * E_ * 2);
  float*    scores = (float*)carve((size_t)B_ * S_ * S_ * 4);
  float*    wpart  = (float*)carve((size_t)B_ * 32 * S_ * 4);
  float*    wsum   = (float*)carve((size_t)B_ * S_ * 4);
  float*    vout   = (float*)carve((size_t)B_ * E_ * 4);
  float*    hbuf   = (float*)carve((size_t)B_ * H_ * 4);
  float*    logits = (float*)carve((size_t)B_ * V_ * 4);

  // 1) gather embeddings -> f16, convert projection weights -> f16
  k_gather<<<B_ * S_, 128, 0, stream>>>(x, emb, e16);
  k_f32_to_f16<<<(E_ * E_) / 256, 256, 0, stream>>>(Wq, wq16, E_ * E_);
  k_f32_to_f16<<<(E_ * E_) / 256, 256, 0, stream>>>(Wk, wk16, E_ * E_);
  k_f32_to_f16<<<(E_ * E_) / 256, 256, 0, stream>>>(Wv, wv16, E_ * E_);

  // 2) q/k/v projections: [16384,512] x [512,512]^T via WMMA (f16 in, f16 out)
  dim3 gqkv((B_ * S_) / 64, E_ / 128, 1);
  k_gemm_abT<true><<<gqkv, 256, 0, stream>>>(e16, wq16, q16, E_, E_, 0, 0, 0);
  k_gemm_abT<true><<<gqkv, 256, 0, stream>>>(e16, wk16, kk16, E_, E_, 0, 0, 0);
  k_gemm_abT<true><<<gqkv, 256, 0, stream>>>(e16, wv16, v16m, E_, E_, 0, 0, 0);

  // 3) scores[b] = q[b] @ k[b]^T via WMMA (f32 out), batched over z
  dim3 gsc(S_ / 64, S_ / 128, B_);
  k_gemm_abT<false><<<gsc, 256, 0, stream>>>(q16, kk16, scores, S_, E_,
                                             (long)S_ * E_, (long)S_ * E_, (long)S_ * S_);

  // 4) column sums of per-row softmax (deterministic partials + reduce)
  k_colsum<<<dim3(32, B_), 256, 0, stream>>>(scores, wpart);
  k_wreduce<<<B_, 256, 0, stream>>>(wpart, wsum);

  // 5) v_out = w @ v   [8,512]
  k_vout<<<B_, 256, 0, stream>>>(wsum, v16m, vout);

  // 6) MLP1 + relu  [8,2048]
  k_mlp1<<<dim3(H_ / 256, B_), 256, 0, stream>>>(vout, W1, b1, hbuf);

  // 7) logits (W2 streamed once, h in LDS)  [8,32000]
  k_logits<<<V_ / 256, 256, 0, stream>>>(hbuf, W2, b2, logits);

  // 8) final softmax -> d_out
  k_softmax<<<B_, 256, 0, stream>>>(logits, (float*)d_out);
}
